// GATRNN_30339648979521
// MI455X (gfx1250) — matrix operand order repaired and checked by reading, compiled, not run on hardware
//
#include <hip/hip_runtime.h>
#include <hip/hip_bf16.h>
#include <cstdint>
#include <cstddef>

// ---------------------------------------------------------------------------
// Model dims
// ---------------------------------------------------------------------------
#define BB    8
#define SSN   32
#define NNODE 64
#define TRI   2016
#define DOUTC 256
#define RR    1024
#define N3    3072
#define BSG   256          // B*S graphs
#define SEQD  16384        // N * DOUT
#define HTC   65536        // per-graph hT elements (1024*64)

typedef __attribute__((ext_vector_type(16))) __bf16 v16bf;
typedef __attribute__((ext_vector_type(8)))  float  v8f;

#define DEV __device__ __forceinline__

DEV unsigned short f2bf(float f) {                 // RNE (epilogue stores)
  unsigned u = __float_as_uint(f);
  return (unsigned short)((u + 0x7FFFu + ((u >> 16) & 1u)) >> 16);
}
DEV float bf2f(unsigned short h) { return __uint_as_float((unsigned)h << 16); }

// cheap packed fp32x2 -> bf16x2 (round-half-up + byte-perm): 3 VALU ops
DEV unsigned pk2(float lo, float hi) {
  unsigned ul = __float_as_uint(lo) + 0x8000u;
  unsigned uh = __float_as_uint(hi) + 0x8000u;
  return __builtin_amdgcn_perm(uh, ul, 0x07060302u);  // {uh[31:16], ul[31:16]}
}

union Frag { unsigned u[8]; v16bf v; };

DEV int kidxA(int j, int hi) { return 2 * j + ((j & 4) ? 8 : 0) + 8 * hi; }
DEV int kidxB(int j, int hi) { return 2 * j + ((j & 4) ? 8 : 0) + 8 * hi; }

DEV v8f wmma_bf16(const Frag& a, const Frag& b, v8f c) {
  return __builtin_amdgcn_wmma_f32_16x16x32_bf16(false, a.v, false, b.v,
                                                 (short)0, c, false, false);
}

// ---------------------------------------------------------------------------
// Converters
// ---------------------------------------------------------------------------
__global__ __launch_bounds__(256) void conv_f2bf_kernel(
    const float* __restrict__ in, unsigned short* __restrict__ out, int n) {
  int t = blockIdx.x * blockDim.x + threadIdx.x;
  if (t < n) out[t] = f2bf(in[t]);
}

// W_gat [2048,1024] fp32 -> Wgt bf16 [c=1024][k=2048]  (K-contiguous for B frags)
__global__ __launch_bounds__(256) void conv_wgt_kernel(
    const float* __restrict__ Wg, unsigned short* __restrict__ Wgt) {
  int t = blockIdx.x * blockDim.x + threadIdx.x;   // 2048*1024
  int k = t >> 10, c = t & 1023;
  Wgt[(size_t)c * 2048 + k] = f2bf(Wg[t]);
}

__global__ __launch_bounds__(256) void init_h_kernel(
    float* __restrict__ hf, unsigned short* __restrict__ hb) {
  int t = blockIdx.x * blockDim.x + threadIdx.x;   // 4*16*1024 = 65536
  hf[t] = 0.f;
  hb[t] = 0;
}

// ---------------------------------------------------------------------------
// Generic K-split reduce: C[t] = sum_ks part[ks*total + t] (+ bias[t % cols])
// ---------------------------------------------------------------------------
__global__ __launch_bounds__(256) void reduce_kernel(
    const float* __restrict__ part, const float* __restrict__ bias,
    float* __restrict__ C, int total, int cols, int ksplit) {
  int t = blockIdx.x * blockDim.x + threadIdx.x;
  if (t >= total) return;
  float s = 0.f;
  for (int ks = 0; ks < ksplit; ++ks) s += part[(size_t)ks * total + t];
  if (bias) s += bias[t % cols];
  C[t] = s;
}

// ---------------------------------------------------------------------------
// he partial: he[bs,c] = sum_k xe_bf16[bs,k] * Wgt[c, 32+k]
// K=2016 split into 4 chunks of 512; wave = (ks, nt); 16 M-accumulators.
// ---------------------------------------------------------------------------
__global__ __launch_bounds__(128) void gemm_he_part_kernel(
    const unsigned short* __restrict__ A,   // [256,2016]
    const unsigned short* __restrict__ Wgt, // [1024,2048]
    float* __restrict__ part) {             // [4][256][1024]
  int wave = (blockIdx.x * blockDim.x + threadIdx.x) >> 5;  // 0..255
  int nt = wave & 63, ks = wave >> 6;
  int lane = threadIdx.x & 31, lr = lane & 15, hi = lane >> 4;
  int n0 = nt * 16;
  int kbeg = ks * 512;
  int kend = (kbeg + 512 < TRI) ? kbeg + 512 : TRI;

  v8f acc[16];
  #pragma unroll
  for (int m = 0; m < 16; ++m)
    #pragma unroll
    for (int r = 0; r < 8; ++r) acc[m][r] = 0.f;

  const unsigned short* brow = Wgt + (size_t)(n0 + lr) * 2048 + 32;
  const unsigned short* abase = A + (size_t)lr * TRI;
  for (int k0 = kbeg; k0 < kend; k0 += 32) {
    Frag b;
    #pragma unroll
    for (int j = 0; j < 8; ++j)
      b.u[j] = *(const unsigned*)(brow + k0 + kidxB(j, hi));
    #pragma unroll
    for (int mt = 0; mt < 16; ++mt) {
      Frag a;
      #pragma unroll
      for (int j = 0; j < 8; ++j)
        a.u[j] = *(const unsigned*)(abase + (size_t)mt * 16 * TRI + k0 + kidxA(j, hi));
      acc[mt] = wmma_bf16(a, b, acc[mt]);
    }
  }
  float* dst = part + (size_t)ks * (256 * 1024);
  #pragma unroll
  for (int mt = 0; mt < 16; ++mt)
    #pragma unroll
    for (int r = 0; r < 8; ++r)
      dst[(size_t)(mt * 16 + r + 8 * hi) * 1024 + n0 + lr] = acc[mt][r];
}

// ---------------------------------------------------------------------------
// node GEMM: hn[m,c] = sum_{f<32} xn[m,f]*Wgt[c,f];  h = hn + he[bs,c]
// store hT bf16 [bs][c][node].  One wave per 16x16 tile (K=32 -> 1 WMMA).
// ---------------------------------------------------------------------------
__global__ __launch_bounds__(256) void gemm_node_kernel(
    const unsigned short* __restrict__ Xn,  // [16384,32]
    const unsigned short* __restrict__ Wgt, // [1024,2048]
    const float* __restrict__ he,           // [256,1024]
    unsigned short* __restrict__ hT) {      // [256][1024][64]
  int wid = (blockIdx.x * blockDim.x + threadIdx.x) >> 5;   // 0..65535
  int nt = wid & 63, mt = wid >> 6;
  int lane = threadIdx.x & 31, lr = lane & 15, hi = lane >> 4;
  Frag a, b;
  #pragma unroll
  for (int j = 0; j < 8; ++j) {
    a.u[j] = *(const unsigned*)(Xn + (size_t)(mt * 16 + lr) * 32 + kidxA(j, hi));
    b.u[j] = *(const unsigned*)(Wgt + (size_t)(nt * 16 + lr) * 2048 + kidxB(j, hi));
  }
  v8f c;
  #pragma unroll
  for (int r = 0; r < 8; ++r) c[r] = 0.f;
  c = wmma_bf16(a, b, c);

  int m0 = mt * 16;
  int bs = m0 >> 6;
  int nodeBase = m0 & 63;
  float hev = he[(size_t)bs * 1024 + nt * 16 + lr];
  unsigned short* dst = hT + (size_t)bs * HTC + (size_t)(nt * 16 + lr) * 64 + nodeBase + 8 * hi;
  #pragma unroll
  for (int r = 0; r < 8; ++r) dst[r] = f2bf(c[r] + hev);
}

// ---------------------------------------------------------------------------
// Attention per graph: a_s/a_d, softmax(leaky_relu), gat = mean_h(alpha@h)+b, relu
// One block (256 thr / 8 waves) per graph bs.
// ---------------------------------------------------------------------------
__global__ __launch_bounds__(256) void attention_kernel(
    const unsigned short* __restrict__ hT,
    const float* __restrict__ att_src,  // [4,256]
    const float* __restrict__ att_dst,
    const float* __restrict__ b_gat,    // [256]
    unsigned short* __restrict__ seq) { // [256,16384] bf16
  __shared__ __align__(16) unsigned short alpha[4][64][64];
  __shared__ float aS[64][4];
  __shared__ float aD[64][4];

  int bs = blockIdx.x;
  const unsigned short* hTb = hT + (size_t)bs * HTC;
  int t = threadIdx.x;

  { // stage 1: per-node per-head attention logits
    int n = t & 63, hd = t >> 6;
    float s1 = 0.f, s2 = 0.f;
    for (int e = 0; e < DOUTC; ++e) {
      float hv = bf2f(hTb[(size_t)(hd * 256 + e) * 64 + n]);
      s1 += hv * att_src[hd * 256 + e];
      s2 += hv * att_dst[hd * 256 + e];
    }
    aS[n][hd] = s1;
    aD[n][hd] = s2;
  }
  __syncthreads();

  { // stage 2: softmax over sources (complete graph + self loops => no mask)
    int i = t & 63, hd = t >> 6;
    float ad = aD[i][hd];
    float mx = -1e30f;
    for (int j = 0; j < 64; ++j) {
      float x = ad + aS[j][hd];
      x = (x > 0.f) ? x : 0.2f * x;
      mx = fmaxf(mx, x);
    }
    float sum = 0.f;
    for (int j = 0; j < 64; ++j) {
      float x = ad + aS[j][hd];
      x = (x > 0.f) ? x : 0.2f * x;
      sum += __expf(x - mx);
    }
    float inv = 1.f / sum;
    for (int j = 0; j < 64; ++j) {
      float x = ad + aS[j][hd];
      x = (x > 0.f) ? x : 0.2f * x;
      alpha[hd][i][j] = f2bf(__expf(x - mx) * inv);
    }
  }
  __syncthreads();

  // stage 3: gat[i,e] = 0.25 * sum_hd (alpha_hd @ h_hd)[i,e]; +b_gat; relu
  int wave = t >> 5, lane = t & 31, lr = lane & 15, hi = lane >> 4;
  for (int tt = 0; tt < 8; ++tt) {
    int tile = wave + 8 * tt;       // 0..63
    int mt = tile & 3, nt = tile >> 2;
    v8f acc;
    #pragma unroll
    for (int r = 0; r < 8; ++r) acc[r] = 0.f;
    #pragma unroll
    for (int hd = 0; hd < 4; ++hd) {
      #pragma unroll
      for (int ks = 0; ks < 2; ++ks) {
        int k0 = ks * 32;
        Frag a, b;
        #pragma unroll
        for (int j = 0; j < 8; ++j) {
          a.u[j] = *(const unsigned*)&alpha[hd][mt * 16 + lr][k0 + kidxA(j, hi)];
          b.u[j] = *(const unsigned*)(hTb + (size_t)(hd * 256 + nt * 16 + lr) * 64
                                      + k0 + kidxB(j, hi));
        }
        acc = wmma_bf16(a, b, acc);
      }
    }
    float bg = b_gat[nt * 16 + lr];
    #pragma unroll
    for (int r = 0; r < 8; ++r) {
      float v = 0.25f * acc[r] + bg;
      v = fmaxf(v, 0.f);
      int node = mt * 16 + r + 8 * hi;
      seq[(size_t)bs * SEQD + (size_t)node * 256 + nt * 16 + lr] = f2bf(v);
    }
  }
}

// ---------------------------------------------------------------------------
// gi partial GEMM: part[ks][256][3072] = A_bf16[256,Kchunk] @ W_f32[3072,K]^T
// wave = (ks, nt over 192 strips); 16 M-accumulators; W streamed non-temporal
// (each weight element read exactly once across all waves).
// ---------------------------------------------------------------------------
__global__ __launch_bounds__(128) void gemm_gi_part_kernel(
    const unsigned short* __restrict__ A,
    const float* __restrict__ W,
    float* __restrict__ part,
    int K, int kchunk) {
  int wave = (blockIdx.x * blockDim.x + threadIdx.x) >> 5;
  int nt = wave % 192, ks = wave / 192;
  int lane = threadIdx.x & 31, lr = lane & 15, hi = lane >> 4;
  int n0 = nt * 16;
  int kbeg = ks * kchunk;
  int kend = (kbeg + kchunk < K) ? kbeg + kchunk : K;

  v8f acc[16];
  #pragma unroll
  for (int m = 0; m < 16; ++m)
    #pragma unroll
    for (int r = 0; r < 8; ++r) acc[m][r] = 0.f;

  const float* wrow = W + (size_t)(n0 + lr) * K;
  const unsigned short* abase = A + (size_t)lr * K;
  for (int k0 = kbeg; k0 < kend; k0 += 32) {
    Frag b;
    #pragma unroll
    for (int j = 0; j < 8; ++j) {
      int kk = k0 + kidxB(j, hi);
      float w0 = __builtin_nontemporal_load(wrow + kk);
      float w1 = __builtin_nontemporal_load(wrow + kk + 1);
      b.u[j] = pk2(w0, w1);   // on-the-fly fp32 -> bf16x2
    }
    #pragma unroll
    for (int mt = 0; mt < 16; ++mt) {
      Frag a;
      #pragma unroll
      for (int j = 0; j < 8; ++j)
        a.u[j] = *(const unsigned*)(abase + (size_t)mt * 16 * K + k0 + kidxA(j, hi));
      acc[mt] = wmma_bf16(a, b, acc[mt]);
    }
  }
  float* dst = part + (size_t)ks * (256 * N3);
  #pragma unroll
  for (int mt = 0; mt < 16; ++mt)
    #pragma unroll
    for (int r = 0; r < 8; ++r)
      dst[(size_t)(mt * 16 + r + 8 * hi) * N3 + n0 + lr] = acc[mt][r];
}

// ---------------------------------------------------------------------------
// GRU recurrent step: gh = h_prev @ W_hh^T (+b_hh), gate math, h update.
// M=16 (8 real batch rows, rows 8..15 kept zero), N=3072, K=1024.
// One wave per 16-wide output strip; 3 gate accumulators; all gate B-frags
// loaded before the WMMAs so 12 b128 loads are in flight per k-step.
// ---------------------------------------------------------------------------
__global__ __launch_bounds__(256) void gru_step_kernel(
    const unsigned short* __restrict__ hprev_bf, // [16,1024]
    const float* __restrict__ hprev_f,           // [16,1024]
    const float* __restrict__ Whh,               // [3072,1024]
    const float* __restrict__ bhh,               // [3072]
    const float* __restrict__ gi,                // [256,3072]
    float* __restrict__ hnext_f,
    unsigned short* __restrict__ hnext_bf,
    unsigned short* __restrict__ yout,           // [256,1024] | null
    int s) {
  int wave = (blockIdx.x * blockDim.x + threadIdx.x) >> 5;  // 0..63
  int lane = threadIdx.x & 31, lr = lane & 15, hi = lane >> 4;
  int o0 = wave * 16;

  v8f ar, az, an;
  #pragma unroll
  for (int r = 0; r < 8; ++r) { ar[r] = 0.f; az[r] = 0.f; an[r] = 0.f; }

  const float* w0 = Whh + (size_t)(o0 + lr) * 1024;
  const float* w1 = Whh + (size_t)(1024 + o0 + lr) * 1024;
  const float* w2 = Whh + (size_t)(2048 + o0 + lr) * 1024;

  for (int k0 = 0; k0 < 1024; k0 += 32) {
    Frag a, b0, b1, b2;
    #pragma unroll
    for (int j = 0; j < 8; ++j)
      a.u[j] = *(const unsigned*)(hprev_bf + (size_t)lr * 1024 + k0 + kidxA(j, hi));
    #pragma unroll
    for (int j = 0; j < 8; ++j) { int kk = k0 + kidxB(j, hi); b0.u[j] = pk2(w0[kk], w0[kk + 1]); }
    #pragma unroll
    for (int j = 0; j < 8; ++j) { int kk = k0 + kidxB(j, hi); b1.u[j] = pk2(w1[kk], w1[kk + 1]); }
    #pragma unroll
    for (int j = 0; j < 8; ++j) { int kk = k0 + kidxB(j, hi); b2.u[j] = pk2(w2[kk], w2[kk + 1]); }
    ar = wmma_bf16(a, b0, ar);
    az = wmma_bf16(a, b1, az);
    an = wmma_bf16(a, b2, an);
  }

  if (hi == 0) {   // lanes 0..15 hold batch rows 0..7 in VGPRs 0..7
    int o = o0 + lr;
    float br = bhh[o], bz = bhh[1024 + o], bn = bhh[2048 + o];
    #pragma unroll
    for (int b = 0; b < 8; ++b) {
      size_t gbase = (size_t)(b * SSN + s) * N3;
      float ir = gi[gbase + o];
      float iz = gi[gbase + 1024 + o];
      float in_ = gi[gbase + 2048 + o];
      float gr = ar[b] + br;
      float gz = az[b] + bz;
      float gn = an[b] + bn;
      float r = 1.f / (1.f + __expf(-(ir + gr)));
      float z = 1.f / (1.f + __expf(-(iz + gz)));
      float n = tanhf(in_ + r * gn);
      float hp = hprev_f[(size_t)b * 1024 + o];
      float hn = (1.f - z) * n + z * hp;
      hnext_f[(size_t)b * 1024 + o] = hn;
      hnext_bf[(size_t)b * 1024 + o] = f2bf(hn);
      if (yout) yout[(size_t)(b * SSN + s) * 1024 + o] = f2bf(hn);
    }
  }
}

// ---------------------------------------------------------------------------
// FC: out[b,o] = h1[b,:] @ W_fc[:,o] + b_fc[o]   (tiny: scalar fp32)
// ---------------------------------------------------------------------------
__global__ __launch_bounds__(256) void fc_kernel(
    const float* __restrict__ h1,   // [16,1024], rows 0..7 valid
    const float* __restrict__ Wfc,  // [1024,2016]
    const float* __restrict__ bfc,  // [2016]
    float* __restrict__ out) {      // [8,2016]
  int t = blockIdx.x * blockDim.x + threadIdx.x;
  if (t >= 8 * TRI) return;
  int b = t / TRI, o = t - b * TRI;
  float s = bfc[o];
  const float* hr = h1 + (size_t)b * 1024;
  for (int r = 0; r < 1024; ++r) s += hr[r] * Wfc[(size_t)r * TRI + o];
  out[t] = s;
}

// ---------------------------------------------------------------------------
// Host orchestration
// ---------------------------------------------------------------------------
extern "C" void kernel_launch(void* const* d_in, const int* in_sizes, int n_in,
                              void* d_out, int out_size, void* d_ws, size_t ws_size,
                              hipStream_t stream) {
  const float* x_node = (const float*)d_in[0];
  const float* x_edge = (const float*)d_in[1];
  // d_in[2] edge_index unused: graph is complete + self loops
  const float* W_gat  = (const float*)d_in[3];
  const float* att_s  = (const float*)d_in[4];
  const float* att_d  = (const float*)d_in[5];
  const float* b_gat  = (const float*)d_in[6];
  const float* W_ih0  = (const float*)d_in[7];
  const float* W_hh0  = (const float*)d_in[8];
  const float* b_ih0  = (const float*)d_in[9];
  const float* b_hh0  = (const float*)d_in[10];
  const float* W_ih1  = (const float*)d_in[11];
  const float* W_hh1  = (const float*)d_in[12];
  const float* b_ih1  = (const float*)d_in[13];
  const float* b_hh1  = (const float*)d_in[14];
  const float* W_fc   = (const float*)d_in[15];
  const float* b_fc   = (const float*)d_in[16];
  float* out = (float*)d_out;

  size_t off = 0;
  char* base = (char*)d_ws;
  auto alloc = [&](size_t bytes) -> void* {
    void* p = base + off;
    off = (off + bytes + 255) & ~(size_t)255;
    return p;
  };
  unsigned short* xe_bf = (unsigned short*)alloc((size_t)BSG * TRI * 2);
  unsigned short* xn_bf = (unsigned short*)alloc((size_t)BSG * NNODE * 32 * 2);
  unsigned short* Wgt   = (unsigned short*)alloc((size_t)1024 * 2048 * 2);
  float*          he    = (float*)alloc((size_t)BSG * 1024 * 4);
  unsigned short* hT    = (unsigned short*)alloc((size_t)BSG * HTC * 2);
  unsigned short* seq   = (unsigned short*)alloc((size_t)BSG * SEQD * 2);
  float*          gi0   = (float*)alloc((size_t)BSG * N3 * 4);
  unsigned short* y0    = (unsigned short*)alloc((size_t)BSG * RR * 2);
  float*          gi1   = (float*)alloc((size_t)BSG * N3 * 4);
  float*          hf    = (float*)alloc((size_t)4 * 16 * RR * 4);  // [layer][parity][16][1024]
  unsigned short* hb    = (unsigned short*)alloc((size_t)4 * 16 * RR * 2);
  float*          part  = (float*)alloc((size_t)8 * BSG * N3 * 4); // K-split partials (25MB, reused)

  // 1) converters + state init
  {
    int n1 = BSG * TRI;           // x_edge
    conv_f2bf_kernel<<<(n1 + 255) / 256, 256, 0, stream>>>(x_edge, xe_bf, n1);
    int n2 = BSG * NNODE * 32;    // x_node
    conv_f2bf_kernel<<<(n2 + 255) / 256, 256, 0, stream>>>(x_node, xn_bf, n2);
    conv_wgt_kernel<<<(2048 * 1024) / 256, 256, 0, stream>>>(W_gat, Wgt);
    init_h_kernel<<<256, 256, 0, stream>>>(hf, hb);
  }

  // 2) edge GEMM (K-split 4 -> 256 waves) + reduce, then node GEMM -> hT
  gemm_he_part_kernel<<<64, 128, 0, stream>>>(xe_bf, Wgt, part);
  reduce_kernel<<<(BSG * 1024 + 255) / 256, 256, 0, stream>>>(
      part, nullptr, he, BSG * 1024, 1024, 4);
  gemm_node_kernel<<<8192, 256, 0, stream>>>(xn_bf, Wgt, he, hT);

  // 3) attention per graph -> seq bf16
  attention_kernel<<<BSG, 256, 0, stream>>>(hT, att_s, att_d, b_gat, seq);

  // 4) layer-0 input GEMM (K=16384, split 8 -> 1536 waves) + reduce(+bias)
  gemm_gi_part_kernel<<<384, 128, 0, stream>>>(seq, W_ih0, part, SEQD, SEQD / 8);
  reduce_kernel<<<(BSG * N3 + 255) / 256, 256, 0, stream>>>(
      part, b_ih0, gi0, BSG * N3, N3, 8);

  // 5) layer-0 recurrence
  for (int s = 0; s < SSN; ++s) {
    int pr = s & 1, pw = pr ^ 1;
    gru_step_kernel<<<8, 256, 0, stream>>>(
        hb + (size_t)(0 * 2 + pr) * 16 * RR, hf + (size_t)(0 * 2 + pr) * 16 * RR,
        W_hh0, b_hh0, gi0,
        hf + (size_t)(0 * 2 + pw) * 16 * RR, hb + (size_t)(0 * 2 + pw) * 16 * RR,
        y0, s);
  }

  // 6) layer-1 input GEMM (K=1024, split 2 -> 384 waves) + reduce(+bias)
  gemm_gi_part_kernel<<<96, 128, 0, stream>>>(y0, W_ih1, part, RR, RR / 2);
  reduce_kernel<<<(BSG * N3 + 255) / 256, 256, 0, stream>>>(
      part, b_ih1, gi1, BSG * N3, N3, 2);

  // 7) layer-1 recurrence
  for (int s = 0; s < SSN; ++s) {
    int pr = s & 1, pw = pr ^ 1;
    gru_step_kernel<<<8, 256, 0, stream>>>(
        hb + (size_t)(1 * 2 + pr) * 16 * RR, hf + (size_t)(1 * 2 + pr) * 16 * RR,
        W_hh1, b_hh1, gi1,
        hf + (size_t)(1 * 2 + pw) * 16 * RR, hb + (size_t)(1 * 2 + pw) * 16 * RR,
        nullptr, s);
  }

  // 8) FC from final layer-1 hidden state (parity 0 after 32 steps)
  fc_kernel<<<(8 * TRI + 255) / 256, 256, 0, stream>>>(
      hf + (size_t)(1 * 2 + 0) * 16 * RR, W_fc, b_fc, out);
}
